// SparseTransformerEncoderLayer_31997506355768
// MI455X (gfx1250) — matrix operand order repaired and checked by reading, not compiled, not run on hardware
//
#include <hip/hip_runtime.h>
#include <hip/hip_bf16.h>

// Problem constants
#define B_  2
#define L_  2048
#define D_  1024
#define H_  16
#define FF_ 4096
#define DH_ 64
#define BL_ (B_*L_)
#define SCALE_ 0.125f   // DH^-0.5

typedef __bf16 bf16x16 __attribute__((ext_vector_type(16)));
typedef float  f32x8   __attribute__((ext_vector_type(8)));

union FragB16 { bf16x16 v; unsigned int u[8]; };

// CDNA5 async global->LDS copy path (ASYNCcnt), if the toolchain exposes it.
#if defined(__has_builtin)
#if __has_builtin(__builtin_amdgcn_global_load_async_to_lds_b128) && \
    __has_builtin(__builtin_amdgcn_s_wait_asynccnt)
#define HAVE_ASYNC_LDS 1
#endif
#endif
#ifndef HAVE_ASYNC_LDS
#define HAVE_ASYNC_LDS 0
#endif

#if HAVE_ASYNC_LDS
typedef int v4i_ __attribute__((vector_size(16)));
typedef __attribute__((address_space(1))) v4i_ as1_v4i;   // prints as "__device__" in HIP
typedef __attribute__((address_space(3))) v4i_ as3_v4i;   // prints as "__shared__" in HIP
__device__ __forceinline__ void async_ld_b128(const void* g, void* l) {
  // generic->LDS addrspace lowering on AMDGPU is low-32-bit truncation,
  // so the integer round-trip is exact.
  __builtin_amdgcn_global_load_async_to_lds_b128(
      (as1_v4i*)(unsigned long long)(uintptr_t)g,
      (as3_v4i*)(unsigned int)(uintptr_t)l, 0, 0);
}
#endif

__device__ __forceinline__ unsigned short f2bf(float f) {
  unsigned int u = __float_as_uint(f);
  u += 0x7FFFu + ((u >> 16) & 1u);          // round-to-nearest-even
  return (unsigned short)(u >> 16);
}

// ---------------------------------------------------------------------------
// fp32 -> bf16 bulk convert
// ---------------------------------------------------------------------------
__global__ __launch_bounds__(256) void cvt_f32_to_bf16(
    const float* __restrict__ in, unsigned short* __restrict__ out, int n) {
  int i = blockIdx.x * 256 + threadIdx.x;
  if (i < n) out[i] = f2bf(in[i]);
}

// ---------------------------------------------------------------------------
// Generic bf16 GEMM:  C[M,N] = A[M,K] @ W[N,K]^T  (+bias, opt GELU, opt res)
// 256 threads (8 waves), 128x128 tile per WG, each wave 32x64 via 2x4 WMMA.
// Tile staging uses GLOBAL_LOAD_ASYNC_TO_LDS_B128 (ASYNCcnt) when available,
// double-buffered with one barrier per 32-deep K-step:
//   wait(own tile-i copies) -> barrier -> issue tile-i+1 into other buffer
//   -> WMMA on tile-i (copies overlap compute, no VGPR staging).
// M%128==0, N%128==0, K%32==0 (all shapes here satisfy this).
// ---------------------------------------------------------------------------
template<bool GELU>
__global__ __launch_bounds__(256) void gemm_bf16(
    const unsigned short* __restrict__ A,   // [M,K] bf16
    const unsigned short* __restrict__ W,   // [N,K] bf16
    const float* __restrict__ bias,         // [N]
    const float* __restrict__ res,          // [M,N] fp32 or nullptr
    float* __restrict__ outf,               // [M,N] fp32 or nullptr
    unsigned short* __restrict__ outb,      // [M,N] bf16 or nullptr
    int M, int N, int K)
{
  constexpr int BM = 128, BN = 128, KT = 32, LDT = 40;
  __shared__ unsigned short As[2][BM * LDT];   // 2 x 10 KB
  __shared__ unsigned short Bs[2][BN * LDT];   // 2 x 10 KB

  const int tid  = threadIdx.x;
  const int lane = tid & 31;
  const int wid  = tid >> 5;
  const int wm   = (wid & 3) * 32;          // 4 waves along M
  const int wn   = (wid >> 2) * 64;         // 2 waves along N (64 wide each)
  const int m0   = blockIdx.y * BM;
  const int n0   = blockIdx.x * BN;
  const int half = lane >> 4;
  const int l15  = lane & 15;
  const int nIter = K / KT;

  f32x8 acc[2][4] = {};

#if HAVE_ASYNC_LDS
  auto issue_tile = [&](int k0, int buf) {
    #pragma unroll
    for (int i = 0; i < 2; ++i) {
      int c = tid + i * 256;                // 512 chunks of 8 bf16 per matrix
      int r = c >> 2, cc = (c & 3) * 8;
      async_ld_b128(A + (size_t)(m0 + r) * K + k0 + cc, &As[buf][r * LDT + cc]);
      async_ld_b128(W + (size_t)(n0 + r) * K + k0 + cc, &Bs[buf][r * LDT + cc]);
    }
  };
  issue_tile(0, 0);
#else
  uint4 ra[2], rb[2];                       // staging registers (fallback path)
  auto g_load = [&](int k0) {
    #pragma unroll
    for (int i = 0; i < 2; ++i) {
      int c = tid + i * 256;
      int r = c >> 2, cc = (c & 3) * 8;
      ra[i] = *(const uint4*)(A + (size_t)(m0 + r) * K + k0 + cc);
      rb[i] = *(const uint4*)(W + (size_t)(n0 + r) * K + k0 + cc);
    }
  };
  auto s_store = [&](int buf) {
    #pragma unroll
    for (int i = 0; i < 2; ++i) {
      int c = tid + i * 256;
      int r = c >> 2, cc = (c & 3) * 8;
      unsigned short* da = &As[buf][r * LDT + cc];
      *(uint2*)da       = make_uint2(ra[i].x, ra[i].y);
      *(uint2*)(da + 4) = make_uint2(ra[i].z, ra[i].w);
      unsigned short* db = &Bs[buf][r * LDT + cc];
      *(uint2*)db       = make_uint2(rb[i].x, rb[i].y);
      *(uint2*)(db + 4) = make_uint2(rb[i].z, rb[i].w);
    }
  };
  g_load(0);
  s_store(0);
  __syncthreads();
#endif

  for (int it = 0; it < nIter; ++it) {
    const int cur = it & 1;

#if HAVE_ASYNC_LDS
    __builtin_amdgcn_s_wait_asynccnt(0);    // own tile-it copies complete
    __syncthreads();                        // all waves' tile-it visible; buf^1 free
    if (it + 1 < nIter) issue_tile((it + 1) * KT, cur ^ 1);
#else
    if (it + 1 < nIter) g_load((it + 1) * KT);   // issue early, hide latency
#endif
    if (it + 2 < nIter) {                         // L2 stream prefetch
      __builtin_prefetch(A + (size_t)(m0 + (tid >> 1)) * K + (it + 2) * KT + (tid & 1) * 16, 0, 0);
      __builtin_prefetch(W + (size_t)(n0 + (tid >> 1)) * K + (it + 2) * KT + (tid & 1) * 16, 0, 0);
    }

    // A fragments (16-bit A 16x32 layout: lane m=l15, half h holds K 8h..8h+7 and 16+8h..23+8h)
    FragB16 a[2], b[4];
    #pragma unroll
    for (int sm = 0; sm < 2; ++sm) {
      int row = wm + sm * 16 + l15;
      #pragma unroll
      for (int v = 0; v < 8; ++v) {
        int k = ((v >> 2) << 4) + (half << 3) + ((v & 3) << 1);
        a[sm].u[v] = *(const unsigned int*)&As[cur][row * LDT + k];
      }
    }
    // B fragments (32x16 layout: lane n=l15, half h holds K 8h..8h+7 and 16+8h..23+8h like A)
    #pragma unroll
    for (int sn = 0; sn < 4; ++sn) {
      int row = wn + sn * 16 + l15;
      #pragma unroll
      for (int v = 0; v < 8; ++v) {
        int k = ((v >> 2) << 4) + (half << 3) + ((v & 3) << 1);
        b[sn].u[v] = *(const unsigned int*)&Bs[cur][row * LDT + k];
      }
    }
    #pragma unroll
    for (int sm = 0; sm < 2; ++sm)
      #pragma unroll
      for (int sn = 0; sn < 4; ++sn)
        acc[sm][sn] = __builtin_amdgcn_wmma_f32_16x16x32_bf16(
            false, a[sm].v, false, b[sn].v, (short)0, acc[sm][sn], false, false);

#if !HAVE_ASYNC_LDS
    if (it + 1 < nIter) s_store(cur ^ 1);   // lands after compute; prev readers synced
    __syncthreads();
#endif
  }

  // Epilogue: C layout m = v + 8*half, n = l15
  #pragma unroll
  for (int sm = 0; sm < 2; ++sm) {
    #pragma unroll
    for (int sn = 0; sn < 4; ++sn) {
      #pragma unroll
      for (int v = 0; v < 8; ++v) {
        int m = m0 + wm + sm * 16 + (half << 3) + v;
        int n = n0 + wn + sn * 16 + l15;
        float val = acc[sm][sn][v] + bias[n];
        if (GELU) val = 0.5f * val * (1.0f + erff(val * 0.70710678118f));
        size_t idx = (size_t)m * N + n;
        if (res)  val += res[idx];
        if (outf) outf[idx] = val;
        if (outb) outb[idx] = f2bf(val);
      }
    }
  }
}

// ---------------------------------------------------------------------------
// Fused attention: per WG = (b, h, 16 query rows)
//  step1: z = 0.5 * SCALE * Q Kt   (WMMA, masked)    -> LDS fp32 [16][2048]
//  step2: entmax-1.5 threshold via bisection on tau  -> LDS bf16 [16][2048]
//  step3: ctx = P @ V (WMMA, split-K over 8 waves, ds_add_f32 reduction)
// Dynamic LDS = 16*2048*4 + 16*2048*2 + 16*64*4 = 200704 B (<320KB/WGP)
// ---------------------------------------------------------------------------
__global__ __launch_bounds__(256) void attn_entmax15(
    const unsigned short* __restrict__ Qb,   // [BL, D] bf16 (head-packed)
    const unsigned short* __restrict__ Kb,   // [BL, D] bf16
    const unsigned short* __restrict__ Vb,   // [BL, D] bf16
    const unsigned char* __restrict__ mask,  // [B, L] bool
    unsigned short* __restrict__ Cb)         // [BL, D] bf16 ctx out
{
  extern __shared__ char smem_raw[];
  float*          zs   = (float*)smem_raw;                             // 16*2048
  unsigned short* ps   = (unsigned short*)(smem_raw + 16*2048*4);      // 16*2048
  float*          ctxs = (float*)(smem_raw + 16*2048*4 + 16*2048*2);   // 16*64

  const int tid  = threadIdx.x;
  const int lane = tid & 31;
  const int wid  = tid >> 5;
  const int half = lane >> 4;
  const int l15  = lane & 15;

  const int idx = blockIdx.x;
  const int qb  = idx & 127;            // L/16 = 128 q-blocks
  const int h   = (idx >> 7) & 15;
  const int b   = idx >> 11;
  const int q0  = qb * 16;

  // zero ctx accumulator
  for (int i = tid; i < 16 * 64; i += 256) ctxs[i] = 0.0f;

  // ---- step 1: scores ----
  const int qrow = q0 + l15;
  FragB16 qf[2];
  #pragma unroll
  for (int ks = 0; ks < 2; ++ks)
    #pragma unroll
    for (int v = 0; v < 8; ++v) {
      int d = ks * 32 + ((v >> 2) << 4) + (half << 3) + ((v & 3) << 1);
      qf[ks].u[v] = *(const unsigned int*)(Qb + (size_t)(b * L_ + qrow) * D_ + h * DH_ + d);
    }
  for (int nt = 0; nt < 16; ++nt) {
    const int key0 = wid * 256 + nt * 16;
    f32x8 acc = {};
    #pragma unroll
    for (int ks = 0; ks < 2; ++ks) {
      FragB16 bf;
      #pragma unroll
      for (int v = 0; v < 8; ++v) {
        int d   = ks * 32 + ((v >> 2) << 4) + (half << 3) + ((v & 3) << 1);
        int key = key0 + l15;
        bf.u[v] = *(const unsigned int*)(Kb + (size_t)(b * L_ + key) * D_ + h * DH_ + d);
      }
      acc = __builtin_amdgcn_wmma_f32_16x16x32_bf16(
          false, qf[ks].v, false, bf.v, (short)0, acc, false, false);
    }
    const int key = key0 + l15;
    const float msk = mask[b * L_ + key] ? 1.0f : 0.0f;
    #pragma unroll
    for (int v = 0; v < 8; ++v) {
      int m = (half << 3) + v;
      float s = acc[v] * SCALE_;
      s = msk != 0.0f ? -1e9f : s;
      zs[m * 2048 + key] = 0.5f * s;     // entmax operates on z/2
    }
  }
  __syncthreads();

  // ---- step 2: entmax-1.5 via bisection (2 rows per wave) ----
  #pragma unroll
  for (int rr = 0; rr < 2; ++rr) {
    const int r = wid * 2 + rr;
    const float* zr = zs + r * 2048;
    float zmax = -3.4e38f;
    for (int i = lane; i < 2048; i += 32) zmax = fmaxf(zmax, zr[i]);
    #pragma unroll
    for (int off = 16; off > 0; off >>= 1)
      zmax = fmaxf(zmax, __shfl_xor(zmax, off, 32));
    float lo = zmax - 1.0f, hi = zmax;
    #pragma unroll 1
    for (int it = 0; it < 30; ++it) {
      float tau = 0.5f * (lo + hi);
      float s = 0.0f;
      for (int i = lane; i < 2048; i += 32) {
        float t = zr[i] - tau;
        if (t > 0.0f) s += t * t;
      }
      #pragma unroll
      for (int off = 16; off > 0; off >>= 1) s += __shfl_xor(s, off, 32);
      if (s >= 1.0f) lo = tau; else hi = tau;
    }
    const float tau = 0.5f * (lo + hi);
    for (int i = lane; i < 2048; i += 32) {
      float t = zr[i] - tau;
      t = t > 0.0f ? t : 0.0f;
      ps[r * 2048 + i] = f2bf(t * t);
    }
  }
  __syncthreads();

  // ---- step 3: ctx = P @ V, split-K across waves ----
  f32x8 acc[4] = {};
  const int kc0 = wid * 256;
  for (int step = 0; step < 8; ++step) {
    const int kb = kc0 + step * 32;
    FragB16 a;
    #pragma unroll
    for (int v = 0; v < 8; ++v) {
      int k = kb + ((v >> 2) << 4) + (half << 3) + ((v & 3) << 1);
      a.u[v] = *(const unsigned int*)&ps[l15 * 2048 + k];
    }
    #pragma unroll
    for (int nt = 0; nt < 4; ++nt) {
      FragB16 bf;
      #pragma unroll
      for (int v = 0; v < 8; ++v) {
        int key = kb + ((v >> 2) << 4) + (half << 3) + ((v & 3) << 1);
        int d   = nt * 16 + l15;
        unsigned int loq = Vb[(size_t)(b * L_ + key)     * D_ + h * DH_ + d];
        unsigned int hiq = Vb[(size_t)(b * L_ + key + 1) * D_ + h * DH_ + d];
        bf.u[v] = loq | (hiq << 16);
      }
      acc[nt] = __builtin_amdgcn_wmma_f32_16x16x32_bf16(
          false, a.v, false, bf.v, (short)0, acc[nt], false, false);
    }
  }
  #pragma unroll
  for (int nt = 0; nt < 4; ++nt)
    #pragma unroll
    for (int v = 0; v < 8; ++v) {
      int m = (half << 3) + v;
      int d = nt * 16 + l15;
      atomicAdd(&ctxs[m * 64 + d], acc[nt][v]);   // ds_add_f32
    }
  __syncthreads();

  for (int i = tid; i < 16 * 64; i += 256) {
    int m = i >> 6, d = i & 63;
    Cb[(size_t)(b * L_ + q0 + m) * D_ + h * DH_ + d] = f2bf(ctxs[i]);
  }
}

// ---------------------------------------------------------------------------
// Host orchestration
// ---------------------------------------------------------------------------
extern "C" void kernel_launch(void* const* d_in, const int* in_sizes, int n_in,
                              void* d_out, int out_size, void* d_ws, size_t ws_size,
                              hipStream_t stream) {
  const float* src  = (const float*)d_in[0];
  const unsigned char* mask = (const unsigned char*)d_in[1];
  const float* qw = (const float*)d_in[2];  const float* qb = (const float*)d_in[3];
  const float* kw = (const float*)d_in[4];  const float* kb = (const float*)d_in[5];
  const float* vw = (const float*)d_in[6];  const float* vb = (const float*)d_in[7];
  const float* ow = (const float*)d_in[8];  const float* ob = (const float*)d_in[9];
  const float* w1 = (const float*)d_in[10]; const float* b1 = (const float*)d_in[11];
  const float* w2 = (const float*)d_in[12]; const float* b2 = (const float*)d_in[13];
  float* out = (float*)d_out;

  char* base = (char*)d_ws;
  size_t off = 0;
  auto alloc = [&](size_t bytes) { char* p = base + off; off += (bytes + 255) & ~(size_t)255; return p; };

  unsigned short* Xb  = (unsigned short*)alloc((size_t)BL_ * D_ * 2);
  unsigned short* Wqb = (unsigned short*)alloc((size_t)D_ * D_ * 2);
  unsigned short* Wkb = (unsigned short*)alloc((size_t)D_ * D_ * 2);
  unsigned short* Wvb = (unsigned short*)alloc((size_t)D_ * D_ * 2);
  unsigned short* Wob = (unsigned short*)alloc((size_t)D_ * D_ * 2);
  unsigned short* W1b = (unsigned short*)alloc((size_t)FF_ * D_ * 2);
  unsigned short* W2b = (unsigned short*)alloc((size_t)D_ * FF_ * 2);
  unsigned short* Qb  = (unsigned short*)alloc((size_t)BL_ * D_ * 2);
  unsigned short* Kb  = (unsigned short*)alloc((size_t)BL_ * D_ * 2);
  unsigned short* Vb  = (unsigned short*)alloc((size_t)BL_ * D_ * 2);
  unsigned short* Cb  = (unsigned short*)alloc((size_t)BL_ * D_ * 2);
  float*          Xr  = (float*)alloc((size_t)BL_ * D_ * 4);          // x = src + attn_out (fp32)
  unsigned short* Xrb = (unsigned short*)alloc((size_t)BL_ * D_ * 2);
  unsigned short* Hb  = (unsigned short*)alloc((size_t)BL_ * FF_ * 2);

  auto cvt = [&](const float* in, unsigned short* o, int n) {
    cvt_f32_to_bf16<<<(n + 255) / 256, 256, 0, stream>>>(in, o, n);
  };
  cvt(src, Xb,  BL_ * D_);
  cvt(qw,  Wqb, D_ * D_);
  cvt(kw,  Wkb, D_ * D_);
  cvt(vw,  Wvb, D_ * D_);
  cvt(ow,  Wob, D_ * D_);
  cvt(w1,  W1b, FF_ * D_);
  cvt(w2,  W2b, D_ * FF_);

  // QKV projections (bf16 out only)
  dim3 gD(D_ / 128, BL_ / 128);
  gemm_bf16<false><<<gD, 256, 0, stream>>>(Xb, Wqb, qb, nullptr, nullptr, Qb, BL_, D_, D_);
  gemm_bf16<false><<<gD, 256, 0, stream>>>(Xb, Wkb, kb, nullptr, nullptr, Kb, BL_, D_, D_);
  gemm_bf16<false><<<gD, 256, 0, stream>>>(Xb, Wvb, vb, nullptr, nullptr, Vb, BL_, D_, D_);

  // Fused entmax-1.5 attention
  attn_entmax15<<<B_ * H_ * (L_ / 16), 256, 200704, stream>>>(Qb, Kb, Vb, mask, Cb);

  // O-projection + residual with src -> x (fp32 + bf16)
  gemm_bf16<false><<<gD, 256, 0, stream>>>(Cb, Wob, ob, src, Xr, Xrb, BL_, D_, D_);

  // FF1 with exact GELU -> h (bf16)
  dim3 gF(FF_ / 128, BL_ / 128);
  gemm_bf16<true><<<gF, 256, 0, stream>>>(Xrb, W1b, b1, nullptr, nullptr, Hb, BL_, FF_, D_);

  // FF2 + residual with x -> final output (fp32)
  gemm_bf16<false><<<gD, 256, 0, stream>>>(Hb, W2b, b2, Xr, out, nullptr, BL_, D_, FF_);
}
